// GeometricGAT_58720792871130
// MI455X (gfx1250) — matrix operand order repaired and checked by reading, not compiled, not run on hardware
//
#include <hip/hip_runtime.h>
#include <math.h>

// ---------------------------------------------------------------------------
// GeometricGAT on a ring graph, fused for MI455X (gfx1250, wave32, WMMA).
// A and W are staged into LDS *pre-swizzled into WMMA fragment order* so each
// operand fragment is 2x ds_load_b128 per lane (no scalar u16 gathers).
// ---------------------------------------------------------------------------

typedef __attribute__((ext_vector_type(16))) __bf16         v16bf;
typedef __attribute__((ext_vector_type(16))) unsigned short v16u;
typedef __attribute__((ext_vector_type(8)))  float          v8f;

static constexpr int Bn    = 128;
static constexpr int Nn    = 4096;
static constexpr int FEA   = 64;
static constexpr int HID   = 128;
static constexpr int OUTD  = 64;
static constexpr int HEADS = 4;
static constexpr int C1    = 32;          // HID / HEADS
static constexpr float NEG_SLOPE = 0.2f;

static constexpr int MB   = 128;          // nodes per workgroup
static constexpr int ROWS = MB + 2;       // +1 halo each side = 130
static constexpr int RT   = 144;          // rows padded to 9 tiles of 16
static constexpr int MT   = RT / 16;      // 9 row tiles

// bf16 round-to-nearest-even from f32 (bit pattern)
__device__ __forceinline__ unsigned short f2bf(float f) {
  unsigned int u = __float_as_uint(f);
  unsigned int r = u + 0x7FFFu + ((u >> 16) & 1u);
  return (unsigned short)(r >> 16);
}

// ---- fragment-order swizzle helpers (ISA 7.12.2, wave32) -------------------
// A 16x32 bf16 tile, element (rl, co) -> (lane, e):
//   half=(co>>3)&1, lane=rl+16*half, e=(co&7)|((co&16)>>1)
__device__ __forceinline__ int a_swz(int rl, int co) {
  const int half = (co >> 3) & 1;
  const int e    = (co & 7) | ((co & 16) >> 1);
  return (rl + (half << 4)) * 16 + e;     // offset within 512-elem tile
}
// B 32x16 bf16 tile, element (ko, nl) -> (lane, e):
//   half=(ko>>3)&1, lane=nl+16*half, e=(ko&7)|((ko&16)>>1)
__device__ __forceinline__ int b_swz(int ko, int nl) {
  const int half = (ko >> 3) & 1;
  const int e    = (ko & 7) | ((ko & 16) >> 1);
  return (nl + (half << 4)) * 16 + e;
}

// ===========================================================================
// Kernel 1: hp1 = bf16(cat(x,coords)) @ bf16(W1)   (WMMA, f32 acc, in LDS)
//           4-head ring attention + bias + ReLU -> h (bf16) in d_ws
// ===========================================================================
__global__ void gat_l1_kernel(const float* __restrict__ x,
                              const float* __restrict__ coords,
                              const float* __restrict__ W1,
                              const float* __restrict__ We1,
                              const float* __restrict__ as1,
                              const float* __restrict__ ad1,
                              const float* __restrict__ ae1,
                              const float* __restrict__ b1,
                              unsigned short* __restrict__ hout) {
  constexpr int K1 = 96;                  // K = 66 zero-padded to 3 k-tiles
  constexpr int KT = 3;                   // k tiles
  constexpr int NT = HID / 16;            // 8 col tiles
  extern __shared__ __align__(32) char smem[];
  unsigned short* sAf = (unsigned short*)smem;         // [MT][KT][32][16]
  unsigned short* sWf = sAf + MT * KT * 512;           // [KT][NT][32][16]
  float*          sHP = (float*)(sWf + KT * NT * 512); // [RT][HID]
  float*          sSs = sHP + RT * HID;                // [RT][HEADS]
  float*          sSd = sSs + RT * HEADS;              // [RT][HEADS]

  const int tid  = threadIdx.x;
  const int lane = tid & 31;
  const int wave = tid >> 5;
  const int nblk = Nn / MB;
  const int b    = blockIdx.x / nblk;
  const int n0   = (blockIdx.x % nblk) * MB;

  // ---- stage A = cat(x, coords) rows [n0-1 .. n0+MB], fragment order ------
  for (int idx = tid; idx < RT * K1; idx += blockDim.x) {
    const int r = idx / K1, c = idx - r * K1;
    float v = 0.f;
    if (r < ROWS) {
      const int node = (n0 - 1 + r + Nn) % Nn;
      const size_t base = (size_t)b * Nn + node;
      if      (c < FEA)      v = x[base * FEA + c];
      else if (c == FEA)     v = coords[base * 2 + 0];
      else if (c == FEA + 1) v = coords[base * 2 + 1];
    }
    const int mt = r >> 4, kt = c >> 5;
    sAf[((mt * KT + kt) << 9) + a_swz(r & 15, c & 31)] = f2bf(v);
  }
  // ---- stage W1 (66x128 -> 96x128 zero-padded), fragment order ------------
  for (int idx = tid; idx < K1 * HID; idx += blockDim.x) {
    const int k = idx / HID, n = idx - k * HID;
    const float v = (k < FEA + 2) ? W1[k * HID + n] : 0.f;
    const int kt = k >> 5, nt = n >> 4;
    sWf[((kt * NT + nt) << 9) + b_swz(k & 31, n & 15)] = f2bf(v);
  }
  __syncthreads();

  // ---- GEMM: 9 row-tiles x 8 col-tiles, K = 3 x 32 ----
  const v16u* pA = (const v16u*)sAf;      // one 32B vector per lane per tile
  const v16u* pB = (const v16u*)sWf;
  for (int t = wave; t < MT * NT; t += (int)(blockDim.x >> 5)) {
    const int mt = t / NT, nt = t % NT;
    v8f acc = {};
#pragma unroll
    for (int kt = 0; kt < KT; ++kt) {
      v16bf af = __builtin_bit_cast(v16bf, pA[(mt * KT + kt) * 32 + lane]);
      v16bf bf = __builtin_bit_cast(v16bf, pB[(kt * NT + nt) * 32 + lane]);
      acc = __builtin_amdgcn_wmma_f32_16x16x32_bf16(false, af, false, bf,
                                                    (short)0, acc, false, false);
    }
    const int rb  = mt * 16 + ((lane >> 4) ? 8 : 0);
    const int col = nt * 16 + (lane & 15);
#pragma unroll
    for (int rr = 0; rr < 8; ++rr) sHP[(rb + rr) * HID + col] = acc[rr];
  }
  __syncthreads();

  // ---- per-node attention scores s_src, s_dst for rows 0..129 ----
  for (int idx = tid; idx < ROWS * HEADS; idx += blockDim.x) {
    const int r = idx >> 2, h = idx & 3;
    const float* hp = sHP + r * HID + h * C1;
    float s1 = 0.f, s2 = 0.f;
#pragma unroll 8
    for (int c = 0; c < C1; ++c) {
      const float v = hp[c];
      s1 += v * as1[h * C1 + c];
      s2 += v * ad1[h * C1 + c];
    }
    sSs[r * HEADS + h] = s1;
    sSd[r * HEADS + h] = s2;
  }
  __syncthreads();

  // ---- attention softmax over the 2 incoming ring edges + aggregate ----
  for (int idx = tid; idx < MB * HEADS; idx += blockDim.x) {
    const int jl = (idx >> 2) + 1;        // interior row 1..128
    const int h  = idx & 3;
    const int nodej = (n0 + jl - 1) % Nn;
    const int nodem = (n0 + jl - 2 + Nn) % Nn;
    const int nodep = (n0 + jl) % Nn;
    const size_t cb = (size_t)b * Nn;
    const float cjx = coords[(cb + nodej) * 2], cjy = coords[(cb + nodej) * 2 + 1];
    const float cmx = coords[(cb + nodem) * 2], cmy = coords[(cb + nodem) * 2 + 1];
    const float cpx = coords[(cb + nodep) * 2], cpy = coords[(cb + nodep) * 2 + 1];
    // edge-score projection: eattr . (We1 @ ae1[h])  (3-vector)
    float ce0 = 0.f, ce1 = 0.f, ce2 = 0.f;
#pragma unroll 8
    for (int c = 0; c < C1; ++c) {
      const float a = ae1[h * C1 + c];
      ce0 += We1[0 * HID + h * C1 + c] * a;
      ce1 += We1[1 * HID + h * C1 + c] * a;
      ce2 += We1[2 * HID + h * C1 + c] * a;
    }
    const float dax = cjx - cmx, day = cjy - cmy;          // dst - src
    const float da  = sqrtf(dax * dax + day * day);
    float la = sSs[(jl - 1) * HEADS + h] + sSd[jl * HEADS + h]
             + dax * ce0 + day * ce1 + da * ce2;
    la = (la > 0.f) ? la : NEG_SLOPE * la;
    const float dbx = cjx - cpx, dby = cjy - cpy;
    const float db  = sqrtf(dbx * dbx + dby * dby);
    float lb = sSs[(jl + 1) * HEADS + h] + sSd[jl * HEADS + h]
             + dbx * ce0 + dby * ce1 + db * ce2;
    lb = (lb > 0.f) ? lb : NEG_SLOPE * lb;
    const float m  = fmaxf(la, lb);
    const float wa = __expf(la - m), wb = __expf(lb - m);
    const float inv = 1.f / (wa + wb);
    const float ala = wa * inv, alb = wb * inv;
    const float* hm  = sHP + (jl - 1) * HID + h * C1;
    const float* hpv = sHP + (jl + 1) * HID + h * C1;
    unsigned short* o = hout + ((size_t)b * Nn + nodej) * HID + h * C1;
#pragma unroll 8
    for (int c = 0; c < C1; ++c) {
      float v = ala * hm[c] + alb * hpv[c] + b1[h * C1 + c];
      o[c] = f2bf(fmaxf(v, 0.f));               // ReLU, store bf16
    }
  }
}

// ===========================================================================
// Kernel 2: hp2 = h(bf16) @ bf16(W2), 1-head ring attention + bias -> d_out
// ===========================================================================
__global__ void gat_l2_kernel(const unsigned short* __restrict__ hin,
                              const float* __restrict__ coords,
                              const float* __restrict__ W2,
                              const float* __restrict__ We2,
                              const float* __restrict__ as2,
                              const float* __restrict__ ad2,
                              const float* __restrict__ ae2,
                              const float* __restrict__ b2,
                              float* __restrict__ out) {
  constexpr int K2 = HID;                 // 128 = 4 k-tiles
  constexpr int KT = 4;
  constexpr int NT = OUTD / 16;           // 4 col tiles
  extern __shared__ __align__(32) char smem[];
  unsigned short* sAf = (unsigned short*)smem;         // [MT][KT][32][16]
  unsigned short* sWf = sAf + MT * KT * 512;           // [KT][NT][32][16]
  float*          sHP = (float*)(sWf + KT * NT * 512); // [RT][OUTD]
  float*          sSs = sHP + RT * OUTD;               // [RT]
  float*          sSd = sSs + RT;                      // [RT]

  const int tid  = threadIdx.x;
  const int lane = tid & 31;
  const int wave = tid >> 5;
  const int nblk = Nn / MB;
  const int b    = blockIdx.x / nblk;
  const int n0   = (blockIdx.x % nblk) * MB;

  // ---- stage A = h rows [n0-1 .. n0+MB] (bf16 in ws), fragment order ------
  for (int idx = tid; idx < RT * K2; idx += blockDim.x) {
    const int r = idx >> 7, c = idx & 127;
    unsigned short v = 0;
    if (r < ROWS) {
      const int node = (n0 - 1 + r + Nn) % Nn;
      v = hin[((size_t)b * Nn + node) * HID + c];
    }
    const int mt = r >> 4, kt = c >> 5;
    sAf[((mt * KT + kt) << 9) + a_swz(r & 15, c & 31)] = v;
  }
  // ---- stage W2 (128x64), fragment order ------
  for (int idx = tid; idx < K2 * OUTD; idx += blockDim.x) {
    const int k = idx >> 6, n = idx & 63;
    const int kt = k >> 5, nt = n >> 4;
    sWf[((kt * NT + nt) << 9) + b_swz(k & 31, n & 15)] = f2bf(W2[k * OUTD + n]);
  }
  __syncthreads();

  // ---- GEMM: 9 row-tiles x 4 col-tiles, K = 4 x 32 ----
  const v16u* pA = (const v16u*)sAf;
  const v16u* pB = (const v16u*)sWf;
  for (int t = wave; t < MT * NT; t += (int)(blockDim.x >> 5)) {
    const int mt = t / NT, nt = t % NT;
    v8f acc = {};
#pragma unroll
    for (int kt = 0; kt < KT; ++kt) {
      v16bf af = __builtin_bit_cast(v16bf, pA[(mt * KT + kt) * 32 + lane]);
      v16bf bf = __builtin_bit_cast(v16bf, pB[(kt * NT + nt) * 32 + lane]);
      acc = __builtin_amdgcn_wmma_f32_16x16x32_bf16(false, af, false, bf,
                                                    (short)0, acc, false, false);
    }
    const int rb  = mt * 16 + ((lane >> 4) ? 8 : 0);
    const int col = nt * 16 + (lane & 15);
#pragma unroll
    for (int rr = 0; rr < 8; ++rr) sHP[(rb + rr) * OUTD + col] = acc[rr];
  }
  __syncthreads();

  // ---- scores (single head, C=64) ----
  for (int idx = tid; idx < ROWS; idx += blockDim.x) {
    const float* hp = sHP + idx * OUTD;
    float s1 = 0.f, s2 = 0.f;
#pragma unroll 8
    for (int c = 0; c < OUTD; ++c) {
      const float v = hp[c];
      s1 += v * as2[c];
      s2 += v * ad2[c];
    }
    sSs[idx] = s1;
    sSd[idx] = s2;
  }
  __syncthreads();

  // ---- attention + aggregate -> out (fp32). 2 threads per row (32 cols ea) --
  for (int idx = tid; idx < MB * 2; idx += blockDim.x) {
    const int jl   = (idx >> 1) + 1;
    const int half = idx & 1;
    const int nodej = (n0 + jl - 1) % Nn;
    const int nodem = (n0 + jl - 2 + Nn) % Nn;
    const int nodep = (n0 + jl) % Nn;
    const size_t cb = (size_t)b * Nn;
    const float cjx = coords[(cb + nodej) * 2], cjy = coords[(cb + nodej) * 2 + 1];
    const float cmx = coords[(cb + nodem) * 2], cmy = coords[(cb + nodem) * 2 + 1];
    const float cpx = coords[(cb + nodep) * 2], cpy = coords[(cb + nodep) * 2 + 1];
    float ce0 = 0.f, ce1 = 0.f, ce2 = 0.f;
#pragma unroll 8
    for (int c = 0; c < OUTD; ++c) {
      const float a = ae2[c];
      ce0 += We2[0 * OUTD + c] * a;
      ce1 += We2[1 * OUTD + c] * a;
      ce2 += We2[2 * OUTD + c] * a;
    }
    const float dax = cjx - cmx, day = cjy - cmy;
    const float da  = sqrtf(dax * dax + day * day);
    float la = sSs[jl - 1] + sSd[jl] + dax * ce0 + day * ce1 + da * ce2;
    la = (la > 0.f) ? la : NEG_SLOPE * la;
    const float dbx = cjx - cpx, dby = cjy - cpy;
    const float db  = sqrtf(dbx * dbx + dby * dby);
    float lb = sSs[jl + 1] + sSd[jl] + dbx * ce0 + dby * ce1 + db * ce2;
    lb = (lb > 0.f) ? lb : NEG_SLOPE * lb;
    const float m  = fmaxf(la, lb);
    const float wa = __expf(la - m), wb = __expf(lb - m);
    const float inv = 1.f / (wa + wb);
    const float ala = wa * inv, alb = wb * inv;
    const float* hm  = sHP + (jl - 1) * OUTD + half * 32;
    const float* hpv = sHP + (jl + 1) * OUTD + half * 32;
    float* o = out + ((size_t)b * Nn + nodej) * OUTD + half * 32;
#pragma unroll 8
    for (int c = 0; c < 32; ++c)
      o[c] = ala * hm[c] + alb * hpv[c] + b2[half * 32 + c];
  }
}

// ===========================================================================
extern "C" void kernel_launch(void* const* d_in, const int* in_sizes, int n_in,
                              void* d_out, int out_size, void* d_ws, size_t ws_size,
                              hipStream_t stream) {
  (void)in_sizes; (void)n_in; (void)out_size; (void)ws_size;
  const float* x      = (const float*)d_in[0];
  const float* coords = (const float*)d_in[1];
  const float* W1     = (const float*)d_in[2];
  const float* We1    = (const float*)d_in[3];
  const float* as1    = (const float*)d_in[4];
  const float* ad1    = (const float*)d_in[5];
  const float* ae1    = (const float*)d_in[6];
  const float* b1     = (const float*)d_in[7];
  const float* W2     = (const float*)d_in[8];
  const float* We2    = (const float*)d_in[9];
  const float* as2    = (const float*)d_in[10];
  const float* ad2    = (const float*)d_in[11];
  const float* ae2    = (const float*)d_in[12];
  const float* b2     = (const float*)d_in[13];
  // d_in[14]=src, d_in[15]=dst: ring structure is known statically; unused.

  unsigned short* h_ws = (unsigned short*)d_ws;   // [B][N][HID] bf16, 134 MB

  const int nblocks = Bn * (Nn / MB);             // 4096 workgroups
  const size_t lds1 = (size_t)MT * 3 * 512 * 2 + (size_t)3 * 8 * 512 * 2
                    + (size_t)RT * HID * 4 + (size_t)RT * HEADS * 4 * 2;   // ~130.5 KB
  const size_t lds2 = (size_t)MT * 4 * 512 * 2 + (size_t)4 * 4 * 512 * 2
                    + (size_t)RT * OUTD * 4 + (size_t)RT * 4 * 2;          // ~91.3 KB

  gat_l1_kernel<<<nblocks, 256, lds1, stream>>>(x, coords, W1, We1, as1, ad1,
                                                ae1, b1, h_ws);
  gat_l2_kernel<<<nblocks, 256, lds2, stream>>>(h_ws, coords, W2, We2, as2, ad2,
                                                ae2, b2, (float*)d_out);
}
